// self_attention_56977036149299
// MI455X (gfx1250) — compile-verified, matrix operand order checked by reading, not run on hardware
//
#include <hip/hip_runtime.h>
#include <hip/hip_bf16.h>

// ---------------------------------------------------------------------------
// Self-attention (B=4, S=2048, D=1024, fp32 in/out), CDNA5 / gfx1250.
// All matmuls: v_wmma_f32_16x16x32_bf16, 32x32 output per wave (4 WMMA / step,
// A/B fragment reuse). bf16 GEMMs stage tiles with GLOBAL_LOAD_ASYNC_TO_LDS
// (ASYNCcnt). k_scores defined first so the disasm snippet shows the async path.
// Pipeline: k_qkv -> k_scores -> k_softmax -> k_av
// Workspace: Qb(16MB) Kb(16MB) Vt(16MB, transposed) P(32MB logits/probs)
// ---------------------------------------------------------------------------

typedef __attribute__((ext_vector_type(16))) __bf16 v16bf;
typedef __attribute__((ext_vector_type(8)))  float  v8f;
typedef __attribute__((ext_vector_type(4)))  int    v4i_t;

#define AS1 __attribute__((address_space(1)))
#define AS3 __attribute__((address_space(3)))

struct F8    { float f[8]; };
struct U16x2 { uint4 lo, hi; };   // 16 bf16 = 32 bytes

#define BM 128   // block tile M
#define BN 64    // block tile N
#define BK 32    // K step
#define LDT 40   // LDS row stride in bf16 elems (32 + 8 pad -> conflict-free b128)

#define SEQ 2048
#define DIM 1024

#if __has_builtin(__builtin_amdgcn_global_load_async_to_lds_b128) && \
    __has_builtin(__builtin_amdgcn_s_wait_asynccnt)
#define ASYNC_LDS 1
#else
#define ASYNC_LDS 0
#endif

__device__ __forceinline__ unsigned short f2bf(float f) {
  unsigned u = __float_as_uint(f);
  u += 0x7fffu + ((u >> 16) & 1u);        // round-to-nearest-even
  return (unsigned short)(u >> 16);
}

// copy 16B global -> LDS (async-to-LDS when available; ASYNCcnt-tracked)
__device__ __forceinline__ void cp_b128(const unsigned short* g, unsigned short* l) {
#if ASYNC_LDS
  __builtin_amdgcn_global_load_async_to_lds_b128(
      (AS1 v4i_t*)(unsigned long long)(const void*)g,
      (AS3 v4i_t*)(unsigned)(unsigned long long)(void*)l,
      0, 0);
#else
  *(uint4*)l = *(const uint4*)g;
#endif
}

__device__ __forceinline__ void async_wait() {
#if ASYNC_LDS
  __builtin_amdgcn_s_wait_asynccnt(0);
#endif
}

__device__ __forceinline__ v16bf frag_ld(const unsigned short* rowp, int kb) {
  U16x2 t;
  t.lo = *(const uint4*)(rowp + kb);       // K = kb .. kb+7
  t.hi = *(const uint4*)(rowp + kb + 16);  // K = kb+16 .. kb+23
  return __builtin_bit_cast(v16bf, t);
}

__device__ __forceinline__ v8f wmma_bf16(v16bf a, v16bf b, v8f c) {
  return __builtin_amdgcn_wmma_f32_16x16x32_bf16(
      /*neg_a=*/false, a, /*neg_b=*/false, b,
      /*c_mod=*/(short)0, c, /*reuse_a=*/false, /*reuse_b=*/false);
}

// ---------------------------------------------------------------------------
// Shared bf16 GEMM body: C[128x64] += A[128xK] * B[64xK]^T, async LDS staging.
// ---------------------------------------------------------------------------
__device__ __forceinline__ void bf16_gemm_tile(
    const unsigned short* __restrict__ Arow,   // &A[rowbase][0], ld = kdim
    const unsigned short* __restrict__ Brow,   // &B[colbase][0], ld = kdim
    int kdim,
    unsigned short* As, unsigned short* Bs,
    v8f& acc00, v8f& acc01, v8f& acc10, v8f& acc11)
{
  const int tid  = threadIdx.x;
  const int lane = tid & 31;
  const int wave = tid >> 5;
  const int wm   = wave >> 1;
  const int wn   = wave & 1;

  const int lrowA = tid >> 1, lcolA = (tid & 1) * 16;
  const int lrowB = tid >> 2, lcolB = (tid & 3) * 8;
  const unsigned short* ap = Arow + (size_t)lrowA * kdim + lcolA;
  const unsigned short* bp = Brow + (size_t)lrowB * kdim + lcolB;
  unsigned short* la0 = &As[lrowA * LDT + lcolA];
  unsigned short* la1 = &As[lrowA * LDT + lcolA + 8];
  unsigned short* lb  = &Bs[lrowB * LDT + lcolB];

  const int arow = wm * 32 + (lane & 15);
  const int bcol = wn * 32 + (lane & 15);
  const int kb   = (lane >> 4) * 8;

  for (int k0 = 0; k0 < kdim; k0 += BK) {
    cp_b128(ap,     la0);
    cp_b128(ap + 8, la1);
    cp_b128(bp,     lb);
    __builtin_prefetch(ap + BK, 0, 1);    // global_prefetch_b8 next K-tile
    __builtin_prefetch(bp + BK, 0, 1);
    ap += BK; bp += BK;
    async_wait();
    __syncthreads();

    v16bf fa0 = frag_ld(&As[arow * LDT], kb);
    v16bf fa1 = frag_ld(&As[(arow + 16) * LDT], kb);
    v16bf fb0 = frag_ld(&Bs[bcol * LDT], kb);
    v16bf fb1 = frag_ld(&Bs[(bcol + 16) * LDT], kb);
    acc00 = wmma_bf16(fa0, fb0, acc00);
    acc01 = wmma_bf16(fa0, fb1, acc01);
    acc10 = wmma_bf16(fa1, fb0, acc10);
    acc11 = wmma_bf16(fa1, fb1, acc11);
    __syncthreads();
  }
}

// ---------------------------------------------------------------------------
// Kernel: logits P[b,i,j] = scale * sum_d Qb[b,i,d] * Kb[b,j,d]  (bf16 out)
// Defined FIRST in the module so the disasm snippet shows the async GEMM.
// ---------------------------------------------------------------------------
__global__ __launch_bounds__(256) void k_scores(
    const unsigned short* __restrict__ Qb,
    const unsigned short* __restrict__ Kb,
    unsigned short* __restrict__ P, float scale)
{
  const int b = blockIdx.z;
  __shared__ __align__(16) unsigned short As[BM * LDT];
  __shared__ __align__(16) unsigned short Bs[BN * LDT];

  const int tid = threadIdx.x, lane = tid & 31, wave = tid >> 5;
  const int wm = wave >> 1, wn = wave & 1;
  const int rowbase = blockIdx.y * BM;   // query i
  const int colbase = blockIdx.x * BN;   // key j

  v8f acc00 = {}, acc01 = {}, acc10 = {}, acc11 = {};
  bf16_gemm_tile(Qb + (size_t)(b * SEQ + rowbase) * DIM,
                 Kb + (size_t)(b * SEQ + colbase) * DIM,
                 DIM, As, Bs, acc00, acc01, acc10, acc11);

  F8 r[2][2];
  r[0][0] = __builtin_bit_cast(F8, acc00);
  r[0][1] = __builtin_bit_cast(F8, acc01);
  r[1][0] = __builtin_bit_cast(F8, acc10);
  r[1][1] = __builtin_bit_cast(F8, acc11);
  const int n0 = colbase + wn * 32 + (lane & 15);
#pragma unroll
  for (int mt = 0; mt < 2; ++mt) {
#pragma unroll
    for (int j = 0; j < 8; ++j) {
      const int m    = wm * 32 + mt * 16 + ((lane & 16) ? (j + 8) : j);
      const int grow = rowbase + m;
      unsigned short* dst = P + ((size_t)b * SEQ + grow) * SEQ;
      dst[n0]      = f2bf(scale * r[mt][0].f[j]);
      dst[n0 + 16] = f2bf(scale * r[mt][1].f[j]);
    }
  }
}

// ---------------------------------------------------------------------------
// Kernel: O[b,i,d] = sum_j P[b,i,j] * Vt[b,d,j]   (f32 out)
// ---------------------------------------------------------------------------
__global__ __launch_bounds__(256) void k_av(
    const unsigned short* __restrict__ P,
    const unsigned short* __restrict__ Vt,
    float* __restrict__ out)
{
  const int b = blockIdx.z;
  __shared__ __align__(16) unsigned short As[BM * LDT];
  __shared__ __align__(16) unsigned short Bs[BN * LDT];

  const int tid = threadIdx.x, lane = tid & 31, wave = tid >> 5;
  const int wm = wave >> 1, wn = wave & 1;
  const int rowbase = blockIdx.y * BM;   // query i
  const int colbase = blockIdx.x * BN;   // output dim d

  v8f acc00 = {}, acc01 = {}, acc10 = {}, acc11 = {};
  bf16_gemm_tile(P  + ((size_t)b * SEQ + rowbase) * SEQ,
                 Vt + ((size_t)b * DIM + colbase) * SEQ,
                 SEQ, As, Bs, acc00, acc01, acc10, acc11);

  F8 r[2][2];
  r[0][0] = __builtin_bit_cast(F8, acc00);
  r[0][1] = __builtin_bit_cast(F8, acc01);
  r[1][0] = __builtin_bit_cast(F8, acc10);
  r[1][1] = __builtin_bit_cast(F8, acc11);
  const int n0 = colbase + wn * 32 + (lane & 15);
#pragma unroll
  for (int mt = 0; mt < 2; ++mt) {
#pragma unroll
    for (int j = 0; j < 8; ++j) {
      const int m    = wm * 32 + mt * 16 + ((lane & 16) ? (j + 8) : j);
      const int grow = rowbase + m;
      float* dst = out + ((size_t)b * SEQ + grow) * DIM;
      dst[n0]      = r[mt][0].f[j];
      dst[n0 + 16] = r[mt][1].f[j];
    }
  }
}

// ---------------------------------------------------------------------------
// Kernel: fused QKV projection.  y[r,e] = sum_d x[r,d] * W[e,d] + b[e]
// z = blockIdx.z selects (Wq,bq)->Qb, (Wk,bk)->Kb, (Wv,bv)->Vt (transposed).
// f32 inputs converted to bf16 while staging into LDS (synchronous path).
// ---------------------------------------------------------------------------
__global__ __launch_bounds__(256) void k_qkv(
    const float* __restrict__ x,
    const float* __restrict__ Wq, const float* __restrict__ bq,
    const float* __restrict__ Wk, const float* __restrict__ bk,
    const float* __restrict__ Wv, const float* __restrict__ bv,
    unsigned short* __restrict__ Qb, unsigned short* __restrict__ Kb,
    unsigned short* __restrict__ Vt)
{
  const int z = blockIdx.z;
  const float* __restrict__ W    = (z == 0) ? Wq : (z == 1) ? Wk : Wv;
  const float* __restrict__ bias = (z == 0) ? bq : (z == 1) ? bk : bv;

  __shared__ __align__(16) unsigned short As[BM * LDT];
  __shared__ __align__(16) unsigned short Bs[BN * LDT];

  const int tid  = threadIdx.x;
  const int lane = tid & 31;
  const int wave = tid >> 5;
  const int wm   = wave >> 1;       // 4 waves along M (32 rows each)
  const int wn   = wave & 1;        // 2 waves along N (32 cols each)

  const int rowbase = blockIdx.y * BM;
  const int colbase = blockIdx.x * BN;

  // loaders: A tile 128x32 f32 -> thread loads 16 f32; B tile 64x32 -> 8 f32
  const int lrowA = tid >> 1, lcolA = (tid & 1) * 16;
  const int lrowB = tid >> 2, lcolB = (tid & 3) * 8;
  const float* ap = x + (size_t)(rowbase + lrowA) * DIM + lcolA;
  const float* bp = W + (size_t)(colbase + lrowB) * DIM + lcolB;

  // WMMA fragment coordinates
  const int arow = wm * 32 + (lane & 15);
  const int bcol = wn * 32 + (lane & 15);
  const int kb   = (lane >> 4) * 8;

  // bias -> accumulator init (C layout: lane fixes N, VGPR index fixes M)
  const float bz0 = bias[colbase + wn * 32 + (lane & 15)];
  const float bz1 = bias[colbase + wn * 32 + 16 + (lane & 15)];
  F8 i0, i1;
#pragma unroll
  for (int j = 0; j < 8; ++j) { i0.f[j] = bz0; i1.f[j] = bz1; }
  v8f acc00 = __builtin_bit_cast(v8f, i0);   // (m-tile 0, n-tile 0)
  v8f acc01 = __builtin_bit_cast(v8f, i1);
  v8f acc10 = acc00, acc11 = acc01;

  for (int k0 = 0; k0 < DIM; k0 += BK) {
    float4 a0 = *(const float4*)ap;
    float4 a1 = *(const float4*)(ap + 4);
    float4 a2 = *(const float4*)(ap + 8);
    float4 a3 = *(const float4*)(ap + 12);
    float4 b0 = *(const float4*)bp;
    float4 b1 = *(const float4*)(bp + 4);
    __builtin_prefetch(ap + BK, 0, 1);    // global_prefetch_b8 next K-tile
    __builtin_prefetch(bp + BK, 0, 1);
    ap += BK; bp += BK;

    uint4 pa0, pa1, pb;
    pa0.x = (unsigned)f2bf(a0.x) | ((unsigned)f2bf(a0.y) << 16);
    pa0.y = (unsigned)f2bf(a0.z) | ((unsigned)f2bf(a0.w) << 16);
    pa0.z = (unsigned)f2bf(a1.x) | ((unsigned)f2bf(a1.y) << 16);
    pa0.w = (unsigned)f2bf(a1.z) | ((unsigned)f2bf(a1.w) << 16);
    pa1.x = (unsigned)f2bf(a2.x) | ((unsigned)f2bf(a2.y) << 16);
    pa1.y = (unsigned)f2bf(a2.z) | ((unsigned)f2bf(a2.w) << 16);
    pa1.z = (unsigned)f2bf(a3.x) | ((unsigned)f2bf(a3.y) << 16);
    pa1.w = (unsigned)f2bf(a3.z) | ((unsigned)f2bf(a3.w) << 16);
    pb.x  = (unsigned)f2bf(b0.x) | ((unsigned)f2bf(b0.y) << 16);
    pb.y  = (unsigned)f2bf(b0.z) | ((unsigned)f2bf(b0.w) << 16);
    pb.z  = (unsigned)f2bf(b1.x) | ((unsigned)f2bf(b1.y) << 16);
    pb.w  = (unsigned)f2bf(b1.z) | ((unsigned)f2bf(b1.w) << 16);
    *(uint4*)&As[lrowA * LDT + lcolA]     = pa0;
    *(uint4*)&As[lrowA * LDT + lcolA + 8] = pa1;
    *(uint4*)&Bs[lrowB * LDT + lcolB]     = pb;
    __syncthreads();

    v16bf fa0 = frag_ld(&As[arow * LDT], kb);
    v16bf fa1 = frag_ld(&As[(arow + 16) * LDT], kb);
    v16bf fb0 = frag_ld(&Bs[bcol * LDT], kb);
    v16bf fb1 = frag_ld(&Bs[(bcol + 16) * LDT], kb);
    acc00 = wmma_bf16(fa0, fb0, acc00);
    acc01 = wmma_bf16(fa0, fb1, acc01);
    acc10 = wmma_bf16(fa1, fb0, acc10);
    acc11 = wmma_bf16(fa1, fb1, acc11);
    __syncthreads();
  }

  F8 r[2][2];
  r[0][0] = __builtin_bit_cast(F8, acc00);
  r[0][1] = __builtin_bit_cast(F8, acc01);
  r[1][0] = __builtin_bit_cast(F8, acc10);
  r[1][1] = __builtin_bit_cast(F8, acc11);
  const int n0 = colbase + wn * 32 + (lane & 15);
#pragma unroll
  for (int mt = 0; mt < 2; ++mt) {
#pragma unroll
    for (int j = 0; j < 8; ++j) {
      const int m    = wm * 32 + mt * 16 + ((lane & 16) ? (j + 8) : j);
      const int grow = rowbase + m;
      const unsigned short q0 = f2bf(r[mt][0].f[j]);
      const unsigned short q1 = f2bf(r[mt][1].f[j]);
      if (z == 0) {
        Qb[(size_t)grow * DIM + n0]      = q0;
        Qb[(size_t)grow * DIM + n0 + 16] = q1;
      } else if (z == 1) {
        Kb[(size_t)grow * DIM + n0]      = q0;
        Kb[(size_t)grow * DIM + n0 + 16] = q1;
      } else {                      // V stored transposed: Vt[b][d][s]
        const int b = grow >> 11, i = grow & (SEQ - 1);
        Vt[((size_t)b * DIM + n0)      * SEQ + i] = q0;
        Vt[((size_t)b * DIM + n0 + 16) * SEQ + i] = q1;
      }
    }
  }
}

// ---------------------------------------------------------------------------
// Kernel: row softmax over 2048 bf16 logits, in-place. 1 block / row.
// ---------------------------------------------------------------------------
__global__ __launch_bounds__(256) void k_softmax(unsigned short* __restrict__ P)
{
  const int tid = threadIdx.x, lane = tid & 31, wave = tid >> 5;
  unsigned short* row = P + (size_t)blockIdx.x * SEQ;

  uint4 raw = *(uint4*)(row + tid * 8);
  float v[8];
  v[0] = __uint_as_float(raw.x << 16);
  v[1] = __uint_as_float(raw.x & 0xffff0000u);
  v[2] = __uint_as_float(raw.y << 16);
  v[3] = __uint_as_float(raw.y & 0xffff0000u);
  v[4] = __uint_as_float(raw.z << 16);
  v[5] = __uint_as_float(raw.z & 0xffff0000u);
  v[6] = __uint_as_float(raw.w << 16);
  v[7] = __uint_as_float(raw.w & 0xffff0000u);

  float m = v[0];
#pragma unroll
  for (int j = 1; j < 8; ++j) m = fmaxf(m, v[j]);
#pragma unroll
  for (int o = 16; o > 0; o >>= 1) m = fmaxf(m, __shfl_xor(m, o, 32));

  __shared__ float red[16];
  if (lane == 0) red[wave] = m;
  __syncthreads();
  float gm = red[0];
#pragma unroll
  for (int j = 1; j < 8; ++j) gm = fmaxf(gm, red[j]);

  float s = 0.f;
#pragma unroll
  for (int j = 0; j < 8; ++j) { v[j] = __expf(v[j] - gm); s += v[j]; }
#pragma unroll
  for (int o = 16; o > 0; o >>= 1) s += __shfl_xor(s, o, 32);
  if (lane == 0) red[8 + wave] = s;
  __syncthreads();
  float gs = 0.f;
#pragma unroll
  for (int j = 0; j < 8; ++j) gs += red[8 + j];
  const float inv = __frcp_rn(gs);

  uint4 outp;
  outp.x = (unsigned)f2bf(v[0] * inv) | ((unsigned)f2bf(v[1] * inv) << 16);
  outp.y = (unsigned)f2bf(v[2] * inv) | ((unsigned)f2bf(v[3] * inv) << 16);
  outp.z = (unsigned)f2bf(v[4] * inv) | ((unsigned)f2bf(v[5] * inv) << 16);
  outp.w = (unsigned)f2bf(v[6] * inv) | ((unsigned)f2bf(v[7] * inv) << 16);
  *(uint4*)(row + tid * 8) = outp;
}

// ---------------------------------------------------------------------------
extern "C" void kernel_launch(void* const* d_in, const int* in_sizes, int n_in,
                              void* d_out, int out_size, void* d_ws, size_t ws_size,
                              hipStream_t stream) {
  (void)in_sizes; (void)n_in; (void)out_size; (void)ws_size;

  const float* x  = (const float*)d_in[0];
  const float* Wq = (const float*)d_in[1];
  const float* bq = (const float*)d_in[2];
  const float* Wk = (const float*)d_in[3];
  const float* bk = (const float*)d_in[4];
  const float* Wv = (const float*)d_in[5];
  const float* bv = (const float*)d_in[6];
  float* out = (float*)d_out;

  char* ws = (char*)d_ws;
  const size_t MB = 1024ull * 1024ull;
  unsigned short* Qb = (unsigned short*)(ws + 0 * MB);   // 16 MB: bf16 [B*S][D]
  unsigned short* Kb = (unsigned short*)(ws + 16 * MB);  // 16 MB: bf16 [B*S][D]
  unsigned short* Vt = (unsigned short*)(ws + 32 * MB);  // 16 MB: bf16 [B][D][S]
  unsigned short* P  = (unsigned short*)(ws + 48 * MB);  // 32 MB: bf16 [B][S][S]

  // 1) QKV projections (z: 0=Q, 1=K, 2=V-transposed)
  k_qkv<<<dim3(DIM / BN, (4 * SEQ) / BM, 3), 256, 0, stream>>>(
      x, Wq, bq, Wk, bk, Wv, bv, Qb, Kb, Vt);

  // 2) logits = Q K^T / sqrt(seq_len)  (reference: sqrt(K.shape[1]) == sqrt(2048))
  const float scale = 0.022097086912079608f; // 1/sqrt(2048)
  k_scores<<<dim3(SEQ / BN, SEQ / BM, 4), 256, 0, stream>>>(Qb, Kb, P, scale);

  // 3) row-wise softmax, in place
  k_softmax<<<dim3(4 * SEQ), 256, 0, stream>>>(P);

  // 4) O = P V
  k_av<<<dim3(DIM / BN, SEQ / BM, 4), 256, 0, stream>>>(P, Vt, out);
}
